// CRFAggregator_652835029036
// MI455X (gfx1250) — matrix operand order repaired and checked by reading, not compiled, not run on hardware
//
#include <hip/hip_runtime.h>
#include <hip/hip_bf16.h>

// CRF NLL on gfx1250 (MI455X). Chunked linear-space forward algorithm:
//   M_t[i,j] = exp(T[i,j]) * exp(e_t[j]);  chunk product P_c = prod M_t
//   computed as a chain of 32x32 f16-WMMA products (4x v_wmma_f32_16x16x32_f16
//   per step, one wave per chain, no block barriers), per-step max rescale
//   deferred into the next step's B operand, f32 log-scale accounting.
// Roofline: ~256 MB mandatory traffic => ~11 us @ 23.3 TB/s; 34 GFLOP of
// f16 WMMA is negligible => memory-bound is the target regime.

#define CRF_B 128
#define CRF_S 4096
#define CRF_C 32
#define NCHUNK 32
#define CHUNK_LEN 128
#define NBLK_BT ((CRF_B * CRF_S) / 256)  // 2048 blocks over (b,t)

typedef __attribute__((ext_vector_type(16))) _Float16 v16h;
typedef __attribute__((ext_vector_type(8)))  float    v8f;

__device__ __forceinline__ float wave_max_f(float x) {
#pragma unroll
  for (int off = 16; off >= 1; off >>= 1) x = fmaxf(x, __shfl_xor(x, off, 32));
  return x;
}
__device__ __forceinline__ float wave_sum_f(float x) {
#pragma unroll
  for (int off = 16; off >= 1; off >>= 1) x += __shfl_xor(x, off, 32);
  return x;
}

// ---------------- K0: emissions passthrough copy (out[1..]) ----------------
__global__ void crf_copy_kernel(const float* __restrict__ src, float* __restrict__ dst) {
  const size_t n4 = (size_t)CRF_B * CRF_S * CRF_C / 4;
  for (size_t i = (size_t)blockIdx.x * blockDim.x + threadIdx.x; i < n4;
       i += (size_t)gridDim.x * blockDim.x) {
    float4 q = ((const float4*)src)[i];
    dst[i * 4 + 0] = q.x;
    dst[i * 4 + 1] = q.y;
    dst[i * 4 + 2] = q.z;
    dst[i * 4 + 3] = q.w;
  }
}

// ---------------- K1: argmax(tags) + gold emission partial sums ----------------
__global__ __launch_bounds__(256) void crf_argmax_kernel(
    const float* __restrict__ tags, const float* __restrict__ emis,
    int* __restrict__ tagidx, float* __restrict__ partA) {
  const int idx = blockIdx.x * 256 + threadIdx.x;  // (b,t) flat, < B*S
  const float4* r4 = (const float4*)(tags + (size_t)idx * CRF_C);
  int best = 0;
  float bv = -3.4e38f;
#pragma unroll
  for (int vq = 0; vq < 8; ++vq) {  // strict > keeps first max (jnp.argmax)
    float4 q = r4[vq];
    if (q.x > bv) { bv = q.x; best = vq * 4 + 0; }
    if (q.y > bv) { bv = q.y; best = vq * 4 + 1; }
    if (q.z > bv) { bv = q.z; best = vq * 4 + 2; }
    if (q.w > bv) { bv = q.w; best = vq * 4 + 3; }
  }
  tagidx[idx] = best;
  float gold = emis[(size_t)idx * CRF_C + best];

  __shared__ float red[256];
  red[threadIdx.x] = gold;
  __syncthreads();
  for (int s = 128; s > 0; s >>= 1) {
    if (threadIdx.x < s) red[threadIdx.x] += red[threadIdx.x + s];
    __syncthreads();
  }
  if (threadIdx.x == 0) partA[blockIdx.x] = red[0];
}

// ---------------- K2: start/end/transition path-score partial sums ----------------
__global__ __launch_bounds__(256) void crf_path_kernel(
    const int* __restrict__ tagidx, const float* __restrict__ startT,
    const float* __restrict__ endT, const float* __restrict__ trans,
    float* __restrict__ partB) {
  const int idx = blockIdx.x * 256 + threadIdx.x;
  const int t = idx & (CRF_S - 1);
  const int i = tagidx[idx];
  float s = 0.0f;
  if (t == 0) s += startT[i];
  if (t == CRF_S - 1) s += endT[i];
  else s += trans[i * CRF_C + tagidx[idx + 1]];

  __shared__ float red[256];
  red[threadIdx.x] = s;
  __syncthreads();
  for (int st = 128; st > 0; st >>= 1) {
    if (threadIdx.x < st) red[threadIdx.x] += red[threadIdx.x + st];
    __syncthreads();
  }
  if (threadIdx.x == 0) partB[blockIdx.x] = red[0];
}

// ---------------- K3: one wave per (batch,chunk) 32x32 WMMA product chain ----------------
__global__ __launch_bounds__(128) void crf_chunk_kernel(
    const float* __restrict__ emis, const float* __restrict__ trans,
    float* __restrict__ Pout, float* __restrict__ logOut) {
  __shared__ float    expT[CRF_C * CRF_C];       // exp(transitions), shared by 4 waves
  __shared__ _Float16 Plds[4][CRF_C * CRF_C];    // per-wave running product (f16)

  const int tid = threadIdx.x;
  const int wv = tid >> 5, lane = tid & 31;
  const int hs = lane >> 4;          // half-wave select
  const int l16 = lane & 15;
  const int chain = blockIdx.x * 4 + wv;         // = b*NCHUNK + c
  const int b = chain >> 5;                      // NCHUNK == 32
  const int c = chain & (NCHUNK - 1);
  const int c0 = l16, c1 = l16 + 16;             // columns owned by this lane (B/C/D)

  for (int k = tid; k < CRF_C * CRF_C; k += 128) expT[k] = __expf(trans[k]);
  __syncthreads();  // only block-wide barrier in this kernel

  // exp(T) cached per-lane in WMMA B-layout: elem e -> K = (e<8?e:e+8)+hs*8, N = ni*16+l16
  float expTB[2][16];
#pragma unroll
  for (int ni = 0; ni < 2; ++ni)
#pragma unroll
    for (int e = 0; e < 16; ++e)
      expTB[ni][e] = expT[((e < 8 ? e : e + 8) + hs * 8) * CRF_C + ni * 16 + l16];

  const int t0 = c * CHUNK_LEN + 1;
  const int t1 = (c == NCHUNK - 1) ? (CRF_S - 1) : (c * CHUNK_LEN + CHUNK_LEN);
  const float* erow = emis + (size_t)b * CRF_S * CRF_C;
  _Float16* P = Plds[wv];

  // init P = M_{t0} (raw; entries <= 1.105*e^|e|max, f16-safe), measure its max
  float lm = 0.0f;
  {
    float ee0 = __expf(erow[(size_t)t0 * CRF_C + c0]);
    float ee1 = __expf(erow[(size_t)t0 * CRF_C + c1]);
#pragma unroll 4
    for (int r = 0; r < CRF_C; ++r) {
      float p0 = expT[r * CRF_C + c0] * ee0;
      float p1 = expT[r * CRF_C + c1] * ee1;
      P[r * CRF_C + c0] = (_Float16)p0;
      P[r * CRF_C + c1] = (_Float16)p1;
      lm = fmaxf(lm, fmaxf(p0, p1));
    }
  }
  lm = wave_max_f(lm);
  float inv = 1.0f / lm;     // applied to next step's B; log(m) accounted then
  float logm = __logf(lm);
  float logscale = 0.0f;

  v8f acc[2][2];
#pragma unroll
  for (int mi = 0; mi < 2; ++mi)
#pragma unroll
    for (int ni = 0; ni < 2; ++ni) acc[mi][ni] = (v8f){};

  for (int t = t0 + 1; t <= t1; ++t) {
    __builtin_amdgcn_wave_barrier();  // keep A-loads after prev-iter stores
    // A = running product, ISA A-layout: elem e -> K=(e<8?e:e+8)+hs*8, row = mi*16+l16
    v16h a0, a1;
#pragma unroll
    for (int e = 0; e < 16; ++e) {
      const int k = (e < 8 ? e : e + 8) + hs * 8;
      a0[e] = P[l16 * CRF_C + k];
      a1[e] = P[(16 + l16) * CRF_C + k];
    }
    // emission factors for this lane's two columns; fold in deferred 1/m
    float ee0 = __expf(erow[(size_t)t * CRF_C + c0]) * inv;
    float ee1 = __expf(erow[(size_t)t * CRF_C + c1]) * inv;
    if (t + 4 <= t1) __builtin_prefetch(&erow[(size_t)(t + 4) * CRF_C + lane], 0, 1);
    logscale += logm;  // the deferred scale is applied this step

    v16h b0, b1;
#pragma unroll
    for (int e = 0; e < 16; ++e) {
      b0[e] = (_Float16)(expTB[0][e] * ee0);
      b1[e] = (_Float16)(expTB[1][e] * ee1);
    }
    v8f cz = {};
    acc[0][0] = __builtin_amdgcn_wmma_f32_16x16x32_f16(false, a0, false, b0, (short)0, cz, false, false);
    acc[0][1] = __builtin_amdgcn_wmma_f32_16x16x32_f16(false, a0, false, b1, (short)0, cz, false, false);
    acc[1][0] = __builtin_amdgcn_wmma_f32_16x16x32_f16(false, a1, false, b0, (short)0, cz, false, false);
    acc[1][1] = __builtin_amdgcn_wmma_f32_16x16x32_f16(false, a1, false, b1, (short)0, cz, false, false);

    // one max reduction per full 32x32 step (entries are >= 0)
    lm = 0.0f;
#pragma unroll
    for (int mi = 0; mi < 2; ++mi)
#pragma unroll
      for (int ni = 0; ni < 2; ++ni)
#pragma unroll
        for (int r = 0; r < 8; ++r) lm = fmaxf(lm, acc[mi][ni][r]);
    lm = wave_max_f(lm);
    inv = 1.0f / lm;
    logm = __logf(lm);

    __builtin_amdgcn_wave_barrier();  // keep stores after A-loads
    // store raw acc (f16-safe: <= 32*1.105*e^|e|max); C/D layout: reg r -> row r(+8*hs)
#pragma unroll
    for (int mi = 0; mi < 2; ++mi)
#pragma unroll
      for (int ni = 0; ni < 2; ++ni)
#pragma unroll
        for (int r = 0; r < 8; ++r)
          P[(mi * 16 + r + hs * 8) * CRF_C + ni * 16 + l16] = (_Float16)acc[mi][ni][r];
  }

  // chunk product in f32 (raw; last measured m never applied nor accounted)
  float* myP = Pout + (size_t)chain * (CRF_C * CRF_C);
#pragma unroll
  for (int mi = 0; mi < 2; ++mi)
#pragma unroll
    for (int ni = 0; ni < 2; ++ni)
#pragma unroll
      for (int r = 0; r < 8; ++r)
        myP[(mi * 16 + r + hs * 8) * CRF_C + ni * 16 + l16] = acc[mi][ni][r];
  if (lane == 0) logOut[chain] = logscale;
}

// ---------------- K4: per-batch combine over 32 chunk matrices ----------------
__global__ __launch_bounds__(128) void crf_combine_kernel(
    const float* __restrict__ emis, const float* __restrict__ startT,
    const float* __restrict__ endT, const float* __restrict__ Pmat,
    const float* __restrict__ chunkLog, float* __restrict__ logZ) {
  __shared__ float vbuf[4][CRF_C];
  const int w = threadIdx.x >> 5, lane = threadIdx.x & 31;
  const int b = blockIdx.x * 4 + w;  // one wave per batch

  float a0 = startT[lane] + emis[(size_t)b * CRF_S * CRF_C + lane];
  float m0 = wave_max_f(a0);
  float v = __expf(a0 - m0);
  float logscale = m0;

  for (int c = 0; c < NCHUNK; ++c) {
    vbuf[w][lane] = v;                 // broadcast v via uniform-address LDS reads
    __builtin_amdgcn_wave_barrier();
    const float* P = Pmat + ((size_t)b * NCHUNK + c) * (CRF_C * CRF_C);
    float s0 = 0, s1 = 0, s2 = 0, s3 = 0;
#pragma unroll
    for (int i = 0; i < CRF_C; i += 4) {
      s0 = fmaf(vbuf[w][i + 0], P[(i + 0) * CRF_C + lane], s0);
      s1 = fmaf(vbuf[w][i + 1], P[(i + 1) * CRF_C + lane], s1);
      s2 = fmaf(vbuf[w][i + 2], P[(i + 2) * CRF_C + lane], s2);
      s3 = fmaf(vbuf[w][i + 3], P[(i + 3) * CRF_C + lane], s3);
    }
    float acc = (s0 + s1) + (s2 + s3);
    float mm = wave_max_f(acc);
    v = acc / mm;
    logscale += __logf(mm) + chunkLog[b * NCHUNK + c];
    __builtin_amdgcn_wave_barrier();
  }
  float term = v * __expf(endT[lane]);
  term = wave_sum_f(term);
  if (lane == 0) logZ[b] = logscale + __logf(term);
}

// ---------------- K5: final deterministic reduce -> out[0] = logZ - score ----------------
__global__ __launch_bounds__(256) void crf_final_kernel(
    const float* __restrict__ partA, const float* __restrict__ partB,
    const float* __restrict__ logZ, float* __restrict__ out) {
  float s = 0.0f;
  for (int k = threadIdx.x; k < NBLK_BT; k += 256) s += partA[k] + partB[k];
  float z = 0.0f;
  for (int k = threadIdx.x; k < CRF_B; k += 256) z += logZ[k];
  __shared__ float red[256];
  red[threadIdx.x] = z - s;
  __syncthreads();
  for (int st = 128; st > 0; st >>= 1) {
    if (threadIdx.x < st) red[threadIdx.x] += red[threadIdx.x + st];
    __syncthreads();
  }
  if (threadIdx.x == 0) out[0] = red[0];
}

extern "C" void kernel_launch(void* const* d_in, const int* in_sizes, int n_in,
                              void* d_out, int out_size, void* d_ws, size_t ws_size,
                              hipStream_t stream) {
  const float* emis   = (const float*)d_in[0];
  const float* tags   = (const float*)d_in[1];
  const float* startT = (const float*)d_in[2];
  const float* endT   = (const float*)d_in[3];
  const float* trans  = (const float*)d_in[4];
  float* out = (float*)d_out;

  // workspace layout (~19 MB)
  float* ws       = (float*)d_ws;
  int*   tagidx   = (int*)ws;                                      // B*S ints
  float* Pmat     = ws + (size_t)CRF_B * CRF_S;                    // B*NCHUNK*1024 f32
  float* chunkLog = Pmat + (size_t)CRF_B * NCHUNK * CRF_C * CRF_C; // B*NCHUNK
  float* partA    = chunkLog + (size_t)CRF_B * NCHUNK;             // NBLK_BT
  float* partB    = partA + NBLK_BT;                               // NBLK_BT
  float* logZ     = partB + NBLK_BT;                               // B

  crf_copy_kernel<<<4096, 256, 0, stream>>>(emis, out + 1);
  crf_argmax_kernel<<<NBLK_BT, 256, 0, stream>>>(tags, emis, tagidx, partA);
  crf_path_kernel<<<NBLK_BT, 256, 0, stream>>>(tagidx, startT, endT, trans, partB);
  crf_chunk_kernel<<<(CRF_B * NCHUNK) / 4, 128, 0, stream>>>(emis, trans, Pmat, chunkLog);
  crf_combine_kernel<<<CRF_B / 4, 128, 0, stream>>>(emis, startT, endT, Pmat, chunkLog, logZ);
  crf_final_kernel<<<1, 256, 0, stream>>>(partA, partB, logZ, out);
}
